// GATNet_36971078484058
// MI455X (gfx1250) — matrix operand order repaired and checked by reading, not compiled, not run on hardware
//
#include <hip/hip_runtime.h>
#include <hip/hip_bf16.h>
#include <cstdint>
#include <cstring>

typedef __attribute__((ext_vector_type(16))) __bf16 v16bf;
typedef __attribute__((ext_vector_type(8)))  float  v8f;
typedef __attribute__((ext_vector_type(4)))  float  v4f;

#define NEG_SLOPE 0.2f

__device__ __forceinline__ __bf16 bfc(float f) { return (__bf16)f; }

// order-preserving float<->int map (involution), so atomicMax(int) == float max
__device__ __forceinline__ int enc_f(float f) {
    int i = __float_as_int(f);
    return i ^ ((i >> 31) & 0x7fffffff);
}
__device__ __forceinline__ float dec_f(int e) {
    int i = e ^ ((e >> 31) & 0x7fffffff);
    return __int_as_float(i);
}

__device__ __forceinline__ void get_edge(const int* ei, int64_t E, int64_t e, int& s, int& d) {
    if (e < E) { s = ei[e]; d = ei[E + e]; }
    else       { s = d = (int)(e - E); }   // self loop
}

// ---------------------------------------------------------------- init
__global__ void fill_i32(int* p, int64_t n, int v) {
    int64_t i = (int64_t)blockIdx.x * blockDim.x + threadIdx.x;
    if (i < n) p[i] = v;
}

// ---------------------------------------------------------------- GEMM1: h1[N,64] = x[N,512] @ W1[512,64]  (bf16 WMMA, f32 acc)
// block = 128 threads (4 waves), each wave owns one 16x16 col tile; grid.x = ceil(N/16) row tiles
__global__ void gemm1_wmma(const float* __restrict__ x, const float* __restrict__ W1,
                           float* __restrict__ h1, int N) {
    const int lane = threadIdx.x & 31;
    const int wave = threadIdx.x >> 5;       // 0..3 -> columns [wave*16, wave*16+16)
    const int row_tile = blockIdx.x * 16;
    const int lo = lane & 15;
    const int hi = lane >> 4;                // 0 or 1

    int row = row_tile + lo;                 // A row this lane supplies
    int srow = row < N ? row : (N - 1);      // clamp (keeps EXEC uniform)
    const int c0 = wave * 16;
    const int col = c0 + lo;                 // B column this lane supplies

    v8f acc = {};
    const int khiA = hi * 8;                 // A: K offsets {0..7,16..23} / {8..15,24..31}
    const int khiB = hi * 8;                 // B: K offsets {0..7,16..23} / {8..15,24..31}

    const float* aprow = x + (size_t)srow * 512 + khiA;

#pragma unroll
    for (int kc = 0; kc < 512; kc += 32) {
        // ---- A tile (16x32 bf16): 4 x b128 non-temporal loads (x streamed once; keep L2 for h1/edges)
        const float* ap = aprow + kc;
        v4f a0 = __builtin_nontemporal_load((const v4f*)(ap));
        v4f a1 = __builtin_nontemporal_load((const v4f*)(ap + 4));
        v4f a2 = __builtin_nontemporal_load((const v4f*)(ap + 16));
        v4f a3 = __builtin_nontemporal_load((const v4f*)(ap + 20));
        v16bf A;
        A[0]=bfc(a0[0]); A[1]=bfc(a0[1]); A[2]=bfc(a0[2]); A[3]=bfc(a0[3]);
        A[4]=bfc(a1[0]); A[5]=bfc(a1[1]); A[6]=bfc(a1[2]); A[7]=bfc(a1[3]);
        A[8]=bfc(a2[0]); A[9]=bfc(a2[1]); A[10]=bfc(a2[2]); A[11]=bfc(a2[3]);
        A[12]=bfc(a3[0]); A[13]=bfc(a3[1]); A[14]=bfc(a3[2]); A[15]=bfc(a3[3]);

        // ---- B tile (32x16 bf16): element j -> W1[(kc+khiB+j+(j>=8?8:0)), col]  (W1 is hot in L2)
        const float* bp = W1 + (size_t)(kc + khiB) * 64 + col;
        v16bf B;
#pragma unroll
        for (int j = 0; j < 16; ++j) B[j] = bfc(bp[(size_t)(j + (j >= 8 ? 8 : 0)) * 64]);

        acc = __builtin_amdgcn_wmma_f32_16x16x32_bf16(false, A, false, B,
                                                      (short)0, acc, false, false);
    }

    // ---- store: C/D layout: VGPR r, lanes 0-15 -> M=r, lanes 16-31 -> M=8+r; N = lane&15
    if (row_tile + 16 <= N) {        // uniform fast path: no EXEC churn
#pragma unroll
        for (int r = 0; r < 8; ++r)
            h1[(size_t)(row_tile + hi * 8 + r) * 64 + c0 + lo] = acc[r];
    } else {
#pragma unroll
        for (int r = 0; r < 8; ++r) {
            int m = row_tile + hi * 8 + r;
            if (m < N) h1[(size_t)m * 64 + c0 + lo] = acc[r];
        }
    }
}

// ---------------------------------------------------------------- per-node attention coefficients, layer 1 (8 heads x 8 dims)
__global__ void alpha1_kernel(const float* __restrict__ h1, const float* __restrict__ as,
                              const float* __restrict__ ad, float* __restrict__ as1,
                              float* __restrict__ ad1, int N) {
    int n = blockIdx.x * blockDim.x + threadIdx.x;
    if (n >= N) return;
    const float* row = h1 + (size_t)n * 64;
#pragma unroll
    for (int h = 0; h < 8; ++h) {
        float s = 0.f, d = 0.f;
#pragma unroll
        for (int k = 0; k < 8; ++k) {
            float v = row[h * 8 + k];
            s += v * as[h * 8 + k];
            d += v * ad[h * 8 + k];
        }
        as1[n * 8 + h] = s;
        ad1[n * 8 + h] = d;
    }
}

// ---------------------------------------------------------------- edge pass A (layer1): segment max per (dst, head)
__global__ void edge_max1(const int* __restrict__ ei, int64_t E, int64_t Etot,
                          const float* __restrict__ as1, const float* __restrict__ ad1,
                          int* __restrict__ nmax1) {
    int64_t e = (int64_t)blockIdx.x * blockDim.x + threadIdx.x;
    if (e >= Etot) return;
    int s, d; get_edge(ei, E, e, s, d);
#pragma unroll
    for (int h = 0; h < 8; ++h) {
        float v = as1[(size_t)s * 8 + h] + ad1[(size_t)d * 8 + h];
        v = v > 0.f ? v : NEG_SLOPE * v;
        atomicMax(&nmax1[(size_t)d * 8 + h], enc_f(v));
    }
}

// ---------------------------------------------------------------- edge pass B (layer1): w=exp(e-max); sum += w; acc += w*h[src]
// 64 threads per edge: t = head*8 + dim
__global__ void edge_agg1(const int* __restrict__ ei, int64_t E, int64_t Etot,
                          const float* __restrict__ as1, const float* __restrict__ ad1,
                          const int* __restrict__ nmax1, const float* __restrict__ h1,
                          float* __restrict__ nsum1, float* __restrict__ acc1) {
    int64_t g = (int64_t)blockIdx.x * blockDim.x + threadIdx.x;
    int64_t e = g >> 6;
    int t = (int)(g & 63);
    if (e >= Etot) return;
    int s, d; get_edge(ei, E, e, s, d);
    int h = t >> 3;
    float v = as1[(size_t)s * 8 + h] + ad1[(size_t)d * 8 + h];
    v = v > 0.f ? v : NEG_SLOPE * v;
    float w = __expf(v - dec_f(nmax1[(size_t)d * 8 + h]));
    if ((t & 7) == 0) atomicAdd(&nsum1[(size_t)d * 8 + h], w);
    atomicAdd(&acc1[(size_t)d * 64 + t], w * h1[(size_t)s * 64 + t]);
}

// ---------------------------------------------------------------- node pass (layer1): normalize + bias + ELU (in place)
__global__ void node1(const float* __restrict__ nsum1, const float* __restrict__ b1,
                      float* __restrict__ acc1, int64_t total) {
    int64_t g = (int64_t)blockIdx.x * blockDim.x + threadIdx.x;
    if (g >= total) return;
    int64_t n = g >> 6;
    int t = (int)(g & 63);
    float v = acc1[g] / nsum1[n * 8 + (t >> 3)] + b1[t];
    v = v > 0.f ? v : (__expf(v) - 1.f);
    acc1[g] = v;
}

// ---------------------------------------------------------------- GEMM2 + layer-2 attention coeffs: h2[N,40] = hact @ W2[64,40]
__global__ void gemm2_kernel(const float* __restrict__ hact, const float* __restrict__ W2,
                             const float* __restrict__ atts, const float* __restrict__ attd,
                             float* __restrict__ h2, float* __restrict__ as2,
                             float* __restrict__ ad2, int N) {
    __shared__ float sW[64 * 40];
    __shared__ float sAs[40], sAd[40];
    for (int i = threadIdx.x; i < 64 * 40; i += blockDim.x) sW[i] = W2[i];
    if (threadIdx.x < 40) { sAs[threadIdx.x] = atts[threadIdx.x]; sAd[threadIdx.x] = attd[threadIdx.x]; }
    __syncthreads();
    int n = blockIdx.x * blockDim.x + threadIdx.x;
    if (n >= N) return;
    float row[64];
    const v4f* rp = (const v4f*)(hact + (size_t)n * 64);
#pragma unroll
    for (int i = 0; i < 16; ++i) {
        v4f v = rp[i];
        row[4*i] = v[0]; row[4*i+1] = v[1]; row[4*i+2] = v[2]; row[4*i+3] = v[3];
    }
    float sa = 0.f, da = 0.f;
    for (int c = 0; c < 40; ++c) {
        float s = 0.f;
#pragma unroll
        for (int k = 0; k < 64; ++k) s += row[k] * sW[k * 40 + c];
        h2[(size_t)n * 40 + c] = s;
        sa += s * sAs[c];
        da += s * sAd[c];
    }
    as2[n] = sa;
    ad2[n] = da;
}

// ---------------------------------------------------------------- edge passes, layer 2 (H=1, D=40)
__global__ void edge_max2(const int* __restrict__ ei, int64_t E, int64_t Etot,
                          const float* __restrict__ as2, const float* __restrict__ ad2,
                          int* __restrict__ nmax2) {
    int64_t e = (int64_t)blockIdx.x * blockDim.x + threadIdx.x;
    if (e >= Etot) return;
    int s, d; get_edge(ei, E, e, s, d);
    float v = as2[s] + ad2[d];
    v = v > 0.f ? v : NEG_SLOPE * v;
    atomicMax(&nmax2[d], enc_f(v));
}

__global__ void edge_agg2(const int* __restrict__ ei, int64_t E, int64_t Etot,
                          const float* __restrict__ as2, const float* __restrict__ ad2,
                          const int* __restrict__ nmax2, const float* __restrict__ h2,
                          float* __restrict__ nsum2, float* __restrict__ acc2) {
    int64_t g = (int64_t)blockIdx.x * blockDim.x + threadIdx.x;
    int64_t e = g / 40;
    int c = (int)(g - e * 40);
    if (e >= Etot) return;
    int s, d; get_edge(ei, E, e, s, d);
    float v = as2[s] + ad2[d];
    v = v > 0.f ? v : NEG_SLOPE * v;
    float w = __expf(v - dec_f(nmax2[d]));
    if (c == 0) atomicAdd(&nsum2[d], w);
    atomicAdd(&acc2[(size_t)d * 40 + c], w * h2[(size_t)s * 40 + c]);
}

// ---------------------------------------------------------------- final: normalize + bias + log_softmax(40)
__global__ void final_kernel(const float* __restrict__ acc2, const float* __restrict__ nsum2,
                             const float* __restrict__ b2, float* __restrict__ out, int N) {
    int n = blockIdx.x * blockDim.x + threadIdx.x;
    if (n >= N) return;
    float inv = 1.f / nsum2[n];
    float v[40];
    float m = -3.0e38f;
#pragma unroll
    for (int c = 0; c < 40; ++c) {
        v[c] = acc2[(size_t)n * 40 + c] * inv + b2[c];
        m = v[c] > m ? v[c] : m;
    }
    float se = 0.f;
#pragma unroll
    for (int c = 0; c < 40; ++c) se += __expf(v[c] - m);
    float ls = m + __logf(se);
#pragma unroll
    for (int c = 0; c < 40; ++c) out[(size_t)n * 40 + c] = v[c] - ls;
}

// ----------------------------------------------------------------
static int enc_host(float f) {
    int i; memcpy(&i, &f, 4);
    return i ^ ((i >> 31) & 0x7fffffff);
}

extern "C" void kernel_launch(void* const* d_in, const int* in_sizes, int n_in,
                              void* d_out, int out_size, void* d_ws, size_t ws_size,
                              hipStream_t stream) {
    const float* x     = (const float*)d_in[0];
    const int*   ei    = (const int*)d_in[1];
    const float* W1    = (const float*)d_in[2];
    const float* atts1 = (const float*)d_in[3];
    const float* attd1 = (const float*)d_in[4];
    const float* b1    = (const float*)d_in[5];
    const float* W2    = (const float*)d_in[6];
    const float* atts2 = (const float*)d_in[7];
    const float* attd2 = (const float*)d_in[8];
    const float* b2    = (const float*)d_in[9];
    float* out = (float*)d_out;

    const int64_t N    = in_sizes[0] / 512;
    const int64_t E    = in_sizes[1] / 2;
    const int64_t Etot = E + N;

    // workspace layout (floats). Zero region [nsum1..acc2] and -inf region [nmax1..nmax2] are contiguous.
    float* ws = (float*)d_ws;
    float* h1    = ws; ws += N * 64;
    float* as1   = ws; ws += N * 8;
    float* ad1   = ws; ws += N * 8;
    float* h2    = ws; ws += N * 40;
    float* as2   = ws; ws += N;
    float* ad2   = ws; ws += N;
    int*   nmax1 = (int*)ws; ws += N * 8;
    int*   nmax2 = (int*)ws; ws += N;
    float* nsum1 = ws; ws += N * 8;
    float* acc1  = ws; ws += N * 64;   // later reused in place as ELU activations
    float* nsum2 = ws; ws += N;
    float* acc2  = ws; ws += N * 40;

    const int B = 256;
    auto blocks = [&](int64_t work) { return (unsigned)((work + B - 1) / B); };

    // init accumulators (re-done every call -> graph replay is deterministic)
    (void)hipMemsetAsync(nsum1, 0, (size_t)N * (8 + 64 + 1 + 40) * sizeof(float), stream);
    fill_i32<<<blocks(N * 9), B, 0, stream>>>(nmax1, N * 9, enc_host(-3.0e38f));

    // layer 1
    gemm1_wmma<<<(unsigned)((N + 15) / 16), 128, 0, stream>>>(x, W1, h1, (int)N);
    alpha1_kernel<<<blocks(N), B, 0, stream>>>(h1, atts1, attd1, as1, ad1, (int)N);
    edge_max1<<<blocks(Etot), B, 0, stream>>>(ei, E, Etot, as1, ad1, nmax1);
    edge_agg1<<<blocks(Etot * 64), B, 0, stream>>>(ei, E, Etot, as1, ad1, nmax1, h1, nsum1, acc1);
    node1<<<blocks(N * 64), B, 0, stream>>>(nsum1, b1, acc1, N * 64);

    // layer 2
    gemm2_kernel<<<blocks(N), B, 0, stream>>>(acc1, W2, atts2, attd2, h2, as2, ad2, (int)N);
    edge_max2<<<blocks(Etot), B, 0, stream>>>(ei, E, Etot, as2, ad2, nmax2);
    edge_agg2<<<blocks(Etot * 40), B, 0, stream>>>(ei, E, Etot, as2, ad2, nmax2, h2, nsum2, acc2);
    final_kernel<<<blocks(N), B, 0, stream>>>(acc2, nsum2, b2, out, (int)N);
}
